// Encoder_Block_16853451669916
// MI455X (gfx1250) — compile-verified, matrix operand order checked by reading, not run on hardware
//
#include <hip/hip_runtime.h>
#include <cstdint>
#include <cstddef>

typedef __attribute__((ext_vector_type(16))) __bf16 v16bf;
typedef __attribute__((ext_vector_type(8)))  float  v8f;

using u16 = unsigned short;
using u32 = unsigned int;
using u64 = unsigned long long;

constexpr int Bc = 4;      // batch
constexpr int Nn = 2048;   // sequence
constexpr int Dd = 768;    // model dim
constexpr int Hh = 12;     // heads
constexpr int Kh = 64;     // head dim
constexpr int Md = 3072;   // mlp dim
constexpr int Mr = Bc * Nn;            // 8192 rows
constexpr float ATTN_SCALE = 0.03608439182435161f; // 1/sqrt(768)

// ---------- helpers ----------

__device__ __forceinline__ u16 f2bf(float x) {
  u32 u = __builtin_bit_cast(u32, x);
  u32 r = u + 0x7FFFu + ((u >> 16) & 1u);   // round-to-nearest-even
  return (u16)(r >> 16);
}

union FragA { v16bf v; u32 u[8]; };

// 16x32 bf16 A-fragment (wave32): lane L: m=L%16, half=L/16; VGPR j holds
// K = kbase(j), kbase(j)+1, kbase(j) = (j<4 ? 2j : 16+2(j-4)) + 8*half.
// Dense B-fragment (32x16 K-major) == same loader applied to B^T rows.
__device__ __forceinline__ v16bf load_frag_g(const u16* p, int ld) {
  int lane = threadIdx.x & 31;
  int m = lane & 15, hlf = lane >> 4;
  const u16* row = p + (size_t)m * ld;
  FragA f;
#pragma unroll
  for (int j = 0; j < 8; ++j) {
    int k0 = ((j < 4) ? 2 * j : 16 + 2 * (j - 4)) + 8 * hlf;
    f.u[j] = *reinterpret_cast<const u32*>(row + k0);
  }
  return f.v;
}

// Same fragment load but from an LDS-staged tile with 40-u16 (80B) row pitch.
__device__ __forceinline__ v16bf load_frag_lds(const u16* p) {
  int lane = threadIdx.x & 31;
  int n = lane & 15, hlf = lane >> 4;
  const u16* row = p + n * 40;
  FragA f;
#pragma unroll
  for (int j = 0; j < 8; ++j) {
    int k0 = ((j < 4) ? 2 * j : 16 + 2 * (j - 4)) + 8 * hlf;
    f.u[j] = *reinterpret_cast<const u32*>(row + k0);
  }
  return f.v;
}

__device__ __forceinline__ v8f wmma_bf16(v16bf a, v16bf b, v8f c) {
  return __builtin_amdgcn_wmma_f32_16x16x32_bf16(false, a, false, b, (short)0, c,
                                                 false, false);
}

// ---- 16-lane butterfly reductions on the VALU (v_permlane16_b32), not LDS ----

#if __has_builtin(__builtin_amdgcn_permlane16)
__device__ __forceinline__ float perml16(float v, unsigned lo, unsigned hi) {
  unsigned iv = __builtin_bit_cast(unsigned, v);
  unsigned pv = __builtin_amdgcn_permlane16(iv, iv, lo, hi, false, false);
  return __builtin_bit_cast(float, pv);
}
__device__ __forceinline__ float row_max16(float v) {
  v = fmaxf(v, perml16(v, 0x67452301u, 0xEFCDAB89u));  // xor 1
  v = fmaxf(v, perml16(v, 0x54761032u, 0xDCFE98BAu));  // xor 2
  v = fmaxf(v, perml16(v, 0x32107654u, 0xBA98FEDCu));  // xor 4
  v = fmaxf(v, perml16(v, 0xFEDCBA98u, 0x76543210u));  // xor 8
  return v;
}
__device__ __forceinline__ float row_sum16(float v) {
  v += perml16(v, 0x67452301u, 0xEFCDAB89u);
  v += perml16(v, 0x54761032u, 0xDCFE98BAu);
  v += perml16(v, 0x32107654u, 0xBA98FEDCu);
  v += perml16(v, 0xFEDCBA98u, 0x76543210u);
  return v;
}
#else
__device__ __forceinline__ float row_max16(float v) {
  v = fmaxf(v, __shfl_xor(v, 1, 32));
  v = fmaxf(v, __shfl_xor(v, 2, 32));
  v = fmaxf(v, __shfl_xor(v, 4, 32));
  v = fmaxf(v, __shfl_xor(v, 8, 32));
  return v;
}
__device__ __forceinline__ float row_sum16(float v) {
  v += __shfl_xor(v, 1, 32);
  v += __shfl_xor(v, 2, 32);
  v += __shfl_xor(v, 4, 32);
  v += __shfl_xor(v, 8, 32);
  return v;
}
#endif

// ---------- weight transpose + bf16 cast:  out[c*R + r] = bf16(in[r*C + c]) ----------

__global__ __launch_bounds__(256) void transpose_bf16_kernel(
    const float* __restrict__ in, u16* __restrict__ out, int R, int C) {
  int idx = blockIdx.x * 256 + threadIdx.x;
  if (idx < R * C) {
    int r = idx / C, c = idx % C;
    out[(size_t)c * R + r] = f2bf(in[idx]);
  }
}

// ---------- LayerNorm over the SEQUENCE axis (axis=-2), unbiased std, eps on sigma ----------

__global__ __launch_bounds__(256) void ln_seq_kernel(
    const float* __restrict__ x, const float* __restrict__ w,
    const float* __restrict__ bia, u16* __restrict__ out) {
  int d = blockIdx.x * 64 + threadIdx.x;
  int b = blockIdx.y;
  const float* base = x + (size_t)b * Nn * Dd + d;
  float s = 0.f, ss = 0.f;
  for (int i = threadIdx.y; i < Nn; i += 4) {
    float v = base[(size_t)i * Dd];
    s += v; ss += v * v;
  }
  __shared__ float shs[4][64], shq[4][64], shm[64], shi[64];
  shs[threadIdx.y][threadIdx.x] = s;
  shq[threadIdx.y][threadIdx.x] = ss;
  __syncthreads();
  if (threadIdx.y == 0) {
    float ts = 0.f, tq = 0.f;
#pragma unroll
    for (int j = 0; j < 4; ++j) { ts += shs[j][threadIdx.x]; tq += shq[j][threadIdx.x]; }
    float mu  = ts / (float)Nn;
    float var = fmaxf((tq - (float)Nn * mu * mu) / (float)(Nn - 1), 0.f);
    shm[threadIdx.x] = mu;
    shi[threadIdx.x] = 1.f / (sqrtf(var) + 1e-6f);
  }
  __syncthreads();
  float mu = shm[threadIdx.x], inv = shi[threadIdx.x];
  float gw = w[d], gb = bia[d];
  for (int i = threadIdx.y; i < Nn; i += 4) {
    float v = (base[(size_t)i * Dd] - mu) * inv * gw + gb;
    out[((size_t)b * Nn + i) * Dd + d] = f2bf(v);
  }
}

// ---------- WMMA GEMM: C[M x Nd] = epi(A[M x Kd] @ Bt[Nd x Kd]^T + bias) ----------
// Block 256 thr = 8 waves; tile 128x128; wave -> 32x64 (2x4 WMMA tiles).
// B tile staged in LDS via async-to-LDS (double buffered, ASYNCcnt tracked).
// EPI: 0 = bf16 store, 1 = bf16 store transposed to [B,H,64,N] (V^T),
//      2 = fp32 store with residual add, 3 = exact GELU -> bf16 store.

template <int EPI>
__global__ __launch_bounds__(256) void gemm_bf16_kernel(
    const u16* __restrict__ A, const u16* __restrict__ Bt,
    const float* __restrict__ bias, u16* __restrict__ outb,
    float* outf, const float* resid, int M, int Nd, int Kd) {
  int tid = threadIdx.x;
  int lane = tid & 31, wid = tid >> 5;
  int wm = wid & 3, wn = wid >> 2;
  int m_base = blockIdx.x * 128 + wm * 32;
  int nblk = blockIdx.y * 128;
  int n_base = nblk + wn * 64;
  (void)M;

  __shared__ u16 bsm[2][128][40];   // 128 rows x 32 k (bf16), 80B pitch, x2 buffers
  const u32 lds0 = (u32)(uintptr_t)&bsm[0][0][0];
  constexpr u32 LDS_STRIDE = 128u * 80u;

  // Stage Bt rows [nblk, nblk+128) x k slice [k0, k0+32) into LDS buffer `buf`.
  // 512 x 16B chunks, 2 per thread, one async-to-LDS b128 per chunk.
  auto stage = [&](int buf, int k0) {
    int c = tid;
#pragma unroll
    for (int it = 0; it < 2; ++it, c += 256) {
      int row = c >> 2, part = c & 3;
      u32 loff = lds0 + (u32)buf * LDS_STRIDE + (u32)row * 80u + (u32)part * 16u;
      const u16* g = Bt + (size_t)(nblk + row) * Kd + k0 + part * 8;
      u64 ga = (u64)(uintptr_t)g;
      asm volatile("global_load_async_to_lds_b128 %0, %1, off"
                   :: "v"(loff), "v"(ga) : "memory");
    }
  };

  stage(0, 0);
  asm volatile("s_wait_asynccnt 0x0" ::: "memory");
  __syncthreads();

  v8f acc[2][4] = {};
  for (int k0 = 0; k0 < Kd; k0 += 32) {
    int cur = (k0 >> 5) & 1;
    if (k0 + 32 < Kd) stage(1 - cur, k0 + 32);   // prefetch next tile (uniform branch)

    __builtin_prefetch(A + (size_t)m_base * Kd + k0 + 256, 0, 0);
    v16bf a0 = load_frag_g(A + (size_t)m_base * Kd + k0, Kd);
    v16bf a1 = load_frag_g(A + (size_t)(m_base + 16) * Kd + k0, Kd);
    v16bf bfr[4];
#pragma unroll
    for (int j = 0; j < 4; ++j)
      bfr[j] = load_frag_lds(&bsm[cur][wn * 64 + j * 16][0]);
#pragma unroll
    for (int j = 0; j < 4; ++j) {
      acc[0][j] = wmma_bf16(a0, bfr[j], acc[0][j]);
      acc[1][j] = wmma_bf16(a1, bfr[j], acc[1][j]);
    }
    asm volatile("s_wait_asynccnt 0x0" ::: "memory");
    __syncthreads();
  }

  int hlf = lane >> 4, l15 = lane & 15;
#pragma unroll
  for (int i = 0; i < 2; ++i)
#pragma unroll
    for (int j = 0; j < 4; ++j)
#pragma unroll
      for (int r = 0; r < 8; ++r) {
        int row = m_base + i * 16 + r + 8 * hlf;   // C layout: VGPR r -> M = r + 8*half
        int col = n_base + j * 16 + l15;           //           lane%16 -> N
        float v = acc[i][j][r] + bias[col];
        if (EPI == 0) {
          outb[(size_t)row * Nd + col] = f2bf(v);
        } else if (EPI == 1) {        // store V transposed: [B,H,Kh,N]
          int bb = row >> 11, nn = row & (Nn - 1);
          int hh = col >> 6,  kd = col & (Kh - 1);
          outb[(size_t)((bb * Hh + hh) * Kh + kd) * Nn + nn] = f2bf(v);
        } else if (EPI == 2) {        // fp32 residual add
          size_t idx = (size_t)row * Nd + col;
          outf[idx] = resid[idx] + v;
        } else {                       // exact GELU
          float g = 0.5f * v * (1.0f + erff(v * 0.70710678118654752f));
          outb[(size_t)row * Nd + col] = f2bf(g);
        }
      }
}

// ---------- Flash attention: S = QK^T/sqrt(D), online softmax, Y = P V ----------
// grid (Nn/64, Bc*Hh), block 128 (4 waves). Each wave owns 16 query rows.

__global__ __launch_bounds__(128) void flash_attn_kernel(
    const u16* __restrict__ q, const u16* __restrict__ k,
    const u16* __restrict__ vt, u16* __restrict__ y) {
  int lane = threadIdx.x & 31, wid = threadIdx.x >> 5;
  int hlf = lane >> 4, l15 = lane & 15;
  int bh = blockIdx.y, b = bh / Hh, h = bh % Hh;
  int m0 = blockIdx.x * 64 + wid * 16;

  const u16* qbase = q + ((size_t)(b * Nn + m0)) * Dd + h * Kh;
  v16bf aq0 = load_frag_g(qbase, Dd);          // kdim 0..31
  v16bf aq1 = load_frag_g(qbase + 32, Dd);     // kdim 32..63
  const u16* kbase = k + ((size_t)b * Nn) * Dd + h * Kh;
  const u16* vhead = vt + ((size_t)(b * Hh + h) * Kh) * Nn;  // V^T: [Kh x Nn]

  float run_max[8], run_sum[8];
  v8f yacc[4] = {};
#pragma unroll
  for (int r = 0; r < 8; ++r) { run_max[r] = -1e30f; run_sum[r] = 0.f; }

  __shared__ u16 pls[4][16][34];  // per-wave P tile (16x32 bf16), padded rows

  for (int mt = 0; mt < Nn; mt += 32) {
    const u16* kb0 = kbase + (size_t)mt * Dd;
    v16bf bk00 = load_frag_g(kb0, Dd);
    v16bf bk01 = load_frag_g(kb0 + 32, Dd);
    v16bf bk10 = load_frag_g(kb0 + (size_t)16 * Dd, Dd);
    v16bf bk11 = load_frag_g(kb0 + (size_t)16 * Dd + 32, Dd);
    v8f zf = {};
    v8f s0 = wmma_bf16(aq0, bk00, zf); s0 = wmma_bf16(aq1, bk01, s0);
    v8f s1 = wmma_bf16(aq0, bk10, zf); s1 = wmma_bf16(aq1, bk11, s1);

    float corr[8];
#pragma unroll
    for (int r = 0; r < 8; ++r) {
      float a = s0[r] * ATTN_SCALE;
      float c = s1[r] * ATTN_SCALE;
      // row r+8*half lives in VGPR r across the 16 lanes of this half
      float mx = row_max16(fmaxf(a, c));
      float nm = fmaxf(run_max[r], mx);
      corr[r] = __expf(run_max[r] - nm);
      run_max[r] = nm;
      float p0 = __expf(a - nm);
      float p1 = __expf(c - nm);
      run_sum[r] = run_sum[r] * corr[r] + row_sum16(p0 + p1);
      int row = r + 8 * hlf;
      pls[wid][row][l15]      = f2bf(p0);   // C-layout -> LDS (rows of P)
      pls[wid][row][16 + l15] = f2bf(p1);
    }
#pragma unroll
    for (int t = 0; t < 4; ++t)
#pragma unroll
      for (int r = 0; r < 8; ++r) yacc[t][r] *= corr[r];

    __syncthreads();
    // Re-read P as a 16x32 A-fragment (layout transpose through LDS)
    FragA ap;
    {
      const u32* rowp = reinterpret_cast<const u32*>(&pls[wid][l15][0]);
#pragma unroll
      for (int j = 0; j < 8; ++j) {
        int k0 = ((j < 4) ? 2 * j : 16 + 2 * (j - 4)) + 8 * hlf;
        ap.u[j] = rowp[k0 >> 1];
      }
    }
#pragma unroll
    for (int t = 0; t < 4; ++t) {
      v16bf bv = load_frag_g(vhead + (size_t)(t * 16) * Nn + mt, Nn);
      yacc[t] = wmma_bf16(ap.v, bv, yacc[t]);
    }
    __syncthreads();
  }

#pragma unroll
  for (int r = 0; r < 8; ++r) {
    float inv = 1.f / run_sum[r];
    int row = m0 + r + 8 * hlf;
#pragma unroll
    for (int t = 0; t < 4; ++t) {
      int col = h * Kh + t * 16 + l15;
      y[((size_t)(b * Nn + row)) * Dd + col] = f2bf(yacc[t][r] * inv);
    }
  }
}

// ---------- launch ----------

extern "C" void kernel_launch(void* const* d_in, const int* in_sizes, int n_in,
                              void* d_out, int out_size, void* d_ws, size_t ws_size,
                              hipStream_t stream) {
  (void)in_sizes; (void)n_in; (void)out_size; (void)ws_size;
  const float* x    = (const float*)d_in[0];
  const float* ln1w = (const float*)d_in[1];
  const float* ln1b = (const float*)d_in[2];
  const float* ln2w = (const float*)d_in[3];
  const float* ln2b = (const float*)d_in[4];
  const float* wq   = (const float*)d_in[5];
  const float* bq   = (const float*)d_in[6];
  const float* wk   = (const float*)d_in[7];
  const float* bk   = (const float*)d_in[8];
  const float* wv   = (const float*)d_in[9];
  const float* bv   = (const float*)d_in[10];
  const float* wo   = (const float*)d_in[11];
  const float* bo   = (const float*)d_in[12];
  const float* w1   = (const float*)d_in[13];
  const float* b1   = (const float*)d_in[14];
  const float* w2   = (const float*)d_in[15];
  const float* b2   = (const float*)d_in[16];

  u16* ws = (u16*)d_ws;
  const size_t SZ = (size_t)Mr * Dd;        // 6291456 elements
  u16* xn  = ws;                            // LN output (reused for both LNs)
  u16* qb  = ws + SZ;
  u16* kb  = ws + 2 * SZ;
  u16* vtb = ws + 3 * SZ;
  u16* yb  = ws + 4 * SZ;
  u16* hb  = qb;                            // q..y region (4*SZ == Mr*Md) reused for MLP hidden
  u16* wqT = ws + 5 * SZ;
  u16* wkT = wqT + (size_t)Dd * Dd;
  u16* wvT = wkT + (size_t)Dd * Dd;
  u16* woT = wvT + (size_t)Dd * Dd;
  u16* w1T = woT + (size_t)Dd * Dd;         // [Md x Dd]
  u16* w2T = w1T + (size_t)Dd * Md;         // [Dd x Md]

  int ne = Dd * Dd;
  transpose_bf16_kernel<<<(ne + 255) / 256, 256, 0, stream>>>(wq, wqT, Dd, Dd);
  transpose_bf16_kernel<<<(ne + 255) / 256, 256, 0, stream>>>(wk, wkT, Dd, Dd);
  transpose_bf16_kernel<<<(ne + 255) / 256, 256, 0, stream>>>(wv, wvT, Dd, Dd);
  transpose_bf16_kernel<<<(ne + 255) / 256, 256, 0, stream>>>(wo, woT, Dd, Dd);
  int ne2 = Dd * Md;
  transpose_bf16_kernel<<<(ne2 + 255) / 256, 256, 0, stream>>>(w1, w1T, Dd, Md);
  transpose_bf16_kernel<<<(ne2 + 255) / 256, 256, 0, stream>>>(w2, w2T, Md, Dd);

  // LN1 (over sequence axis)
  ln_seq_kernel<<<dim3(Dd / 64, Bc), dim3(64, 4), 0, stream>>>(x, ln1w, ln1b, xn);

  dim3 g1(Mr / 128, Dd / 128);
  // Q, K, V projections (V stored transposed per head)
  gemm_bf16_kernel<0><<<g1, 256, 0, stream>>>(xn, wqT, bq, qb,  nullptr, nullptr, Mr, Dd, Dd);
  gemm_bf16_kernel<0><<<g1, 256, 0, stream>>>(xn, wkT, bk, kb,  nullptr, nullptr, Mr, Dd, Dd);
  gemm_bf16_kernel<1><<<g1, 256, 0, stream>>>(xn, wvT, bv, vtb, nullptr, nullptr, Mr, Dd, Dd);

  // attention
  flash_attn_kernel<<<dim3(Nn / 64, Bc * Hh), 128, 0, stream>>>(qb, kb, vtb, yb);

  // output projection + residual -> d_out (holds x1)
  gemm_bf16_kernel<2><<<g1, 256, 0, stream>>>(yb, woT, bo, nullptr, (float*)d_out, x, Mr, Dd, Dd);

  // LN2
  ln_seq_kernel<<<dim3(Dd / 64, Bc), dim3(64, 4), 0, stream>>>((const float*)d_out, ln2w, ln2b, xn);

  // MLP
  gemm_bf16_kernel<3><<<dim3(Mr / 128, Md / 128), 256, 0, stream>>>(xn, w1T, b1, hb, nullptr, nullptr, Mr, Md, Dd);
  gemm_bf16_kernel<2><<<g1, 256, 0, stream>>>(hb, w2T, b2, nullptr, (float*)d_out, (const float*)d_out, Mr, Dd, Md);
}
